// VisModel_10222022164873
// MI455X (gfx1250) — matrix operand order repaired and checked by reading, not compiled, not run on hardware
//
#include <hip/hip_runtime.h>
#include <hip/hip_bf16.h>
#include <math.h>

// ---------------------------------------------------------------------------
// Types for CDNA5 WMMA
// ---------------------------------------------------------------------------
typedef __attribute__((ext_vector_type(16))) _Float16 v16h;
typedef __attribute__((ext_vector_type(8)))  float    v8f;
typedef __attribute__((ext_vector_type(8)))  _Float16 h8;

#define BSZ   64
#define NPG   1030
#define PROJ  512
#define NNODE (BSZ * NPG)          // 65920

// CDNA5 async memory->LDS copy path (guarded: falls back to load+store)
#if __has_builtin(__builtin_amdgcn_global_load_async_to_lds_b128)
#define USE_ASYNC_LDS 1
typedef int v4i __attribute__((vector_size(16)));
typedef __attribute__((address_space(1))) v4i* v4i_gptr;   // global int4*
typedef __attribute__((address_space(3))) v4i* v4i_lptr;   // LDS int4*
#endif

// ---------------------------------------------------------------------------
// WMMA GEMM:  C = act(A[M,K] @ B[K,N] + bias)
//   A,B f16 row-major. Optional f32 and f16 outputs (ldc leading dim).
//   Row remap: orow = (row / rpb) * NPG + rbase + (row % rpb)   (rpb > 0)
//   Block tile 128x128, BK=32, 8 waves (4 in M x 2 in N), wave tile 32x64.
//   A tile staged via GLOBAL_LOAD_ASYNC_TO_LDS_B128 when available.
//   B tile staged transposed (sBT[n][k]) so fragments are contiguous b128s.
// ---------------------------------------------------------------------------
__global__ __launch_bounds__(256) void wmma_gemm_kernel(
    const _Float16* __restrict__ A, const _Float16* __restrict__ B,
    const float* __restrict__ bias, float* __restrict__ C32,
    _Float16* __restrict__ C16, int M, int N, int K, int act,
    int rpb, int rbase, int ldc)
{
    __shared__ _Float16 sA[128 * 32];   // [m][k]
    __shared__ _Float16 sBT[128 * 32];  // [n][k]  (transposed B tile)

    const int m0   = blockIdx.y * 128;
    const int n0   = blockIdx.x * 128;
    const int tid  = threadIdx.x;
    const int wave = tid >> 5;
    const int lane = tid & 31;
    const int wm   = wave & 3;     // wave row   (4 waves cover 128 rows)
    const int wn   = wave >> 2;    // wave col   (2 waves cover 128 cols)
    const int hlf  = lane >> 4;    // lane half  (0: lanes 0-15, 1: 16-31)
    const int ml   = lane & 15;

    const bool fullM = (m0 + 128 <= M);

    v8f acc[2][4] = {};

    for (int k0 = 0; k0 < K; k0 += 32) {
        __syncthreads();

        // ---- stage A tile (128x32 halfs = 512 chunks of 8 halfs) ----
#ifdef USE_ASYNC_LDS
        if (fullM) {
            // direct memory->LDS DMA, tracked by ASYNCcnt
            #pragma unroll
            for (int it = 0; it < 2; ++it) {
                int c   = tid + it * 256;
                int row = c >> 2;
                int ks  = (c & 3) * 8;
                const _Float16* gp = A + (size_t)(m0 + row) * K + k0 + ks;
                const _Float16* lp = sA + row * 32 + ks;
                __builtin_amdgcn_global_load_async_to_lds_b128(
                    (v4i_gptr)(unsigned long long)gp,
                    (v4i_lptr)(unsigned)(unsigned long long)lp, 0, 0);
            }
        } else
#endif
        {
            #pragma unroll
            for (int it = 0; it < 2; ++it) {
                int c   = tid + it * 256;
                int row = c >> 2;
                int ks  = (c & 3) * 8;
                h8 v = {};
                int gr = m0 + row;
                if (gr < M) v = *(const h8*)(A + (size_t)gr * K + k0 + ks);
                *(h8*)(sA + row * 32 + ks) = v;
            }
        }

        // ---- stage B tile transposed: global row-major -> sBT[n][k] ----
        #pragma unroll
        for (int it = 0; it < 2; ++it) {
            int c    = tid + it * 256;
            int krow = c >> 4;
            int nn   = (c & 15) * 8;
            h8 v = *(const h8*)(B + (size_t)(k0 + krow) * N + n0 + nn);
            #pragma unroll
            for (int j = 0; j < 8; ++j)
                sBT[(nn + j) * 32 + krow] = v[j];
        }

        if (k0 + 32 < K) {   // emits global_prefetch_b8 for next tiles
            int gr = m0 + (tid >> 2);
            if (gr < M) __builtin_prefetch(A + (size_t)gr * K + k0 + 32, 0, 0);
            __builtin_prefetch(B + (size_t)(k0 + 32 + (tid >> 4)) * N + n0, 0, 0);
        }

#ifdef USE_ASYNC_LDS
#if __has_builtin(__builtin_amdgcn_s_wait_asynccnt)
        __builtin_amdgcn_s_wait_asynccnt(0);
#else
        asm volatile("s_wait_asynccnt 0" ::: "memory");
#endif
#endif
        __syncthreads();

        // ---- extract fragments (per CDNA5 16-bit WMMA lane layouts) ----
        // A lane (hlf,ml): row r0+ml, k in [8h,8h+8) ++ [16+8h,16+8h+8)
        v16h af[2], bf[4];
        #pragma unroll
        for (int tm = 0; tm < 2; ++tm) {
            const _Float16* pr = sA + (size_t)(wm * 32 + tm * 16 + ml) * 32;
            h8 lo = *(const h8*)(pr + 8 * hlf);
            h8 hi = *(const h8*)(pr + 16 + 8 * hlf);
            af[tm] = __builtin_shufflevector(lo, hi, 0, 1, 2, 3, 4, 5, 6, 7,
                                             8, 9, 10, 11, 12, 13, 14, 15);
        }
        // B lane (hlf,ml): col c0+ml, k in [8h,8h+8) ++ [16+8h,16+8h+8)
        #pragma unroll
        for (int tn = 0; tn < 4; ++tn) {
            const _Float16* pc = sBT + (size_t)(wn * 64 + tn * 16 + ml) * 32
                                     + 8 * hlf;
            h8 lo = *(const h8*)(pc);
            h8 hi = *(const h8*)(pc + 16);
            bf[tn] = __builtin_shufflevector(lo, hi, 0, 1, 2, 3, 4, 5, 6, 7,
                                             8, 9, 10, 11, 12, 13, 14, 15);
        }
        // ---- 8 WMMAs per wave per k-step ----
        #pragma unroll
        for (int tm = 0; tm < 2; ++tm)
            #pragma unroll
            for (int tn = 0; tn < 4; ++tn)
                acc[tm][tn] = __builtin_amdgcn_wmma_f32_16x16x32_f16(
                    false, af[tm], false, bf[tn], (short)0, acc[tm][tn],
                    false, false);
    }

    // ---- epilogue: bias + activation + dual-precision store ----
    #pragma unroll
    for (int tm = 0; tm < 2; ++tm) {
        #pragma unroll
        for (int tn = 0; tn < 4; ++tn) {
            int col = n0 + wn * 64 + tn * 16 + ml;
            #pragma unroll
            for (int r = 0; r < 8; ++r) {
                int row = m0 + wm * 32 + tm * 16 + hlf * 8 + r;
                if (row < M && col < N) {
                    float v = acc[tm][tn][r];
                    if (bias) v += bias[col];
                    if (act == 1) {          // gelu (tanh approx)
                        float x = v;
                        v = 0.5f * x * (1.f + tanhf(0.7978845608f *
                                                    (x + 0.044715f * x * x * x)));
                    } else if (act == 2) {   // relu
                        v = v > 0.f ? v : 0.f;
                    }
                    size_t orow = (size_t)row;
                    if (rpb > 0)
                        orow = (size_t)(row / rpb) * NPG + rbase + (row % rpb);
                    if (C32) C32[orow * ldc + col] = v;
                    if (C16) C16[orow * ldc + col] = (_Float16)v;
                }
            }
        }
    }
}

// ---------------------------------------------------------------------------
// Support kernels
// ---------------------------------------------------------------------------
__global__ void k_f32_to_f16(const float* __restrict__ in,
                             _Float16* __restrict__ out, long n) {
    long i = (long)blockIdx.x * blockDim.x + threadIdx.x;
    if (i < n) out[i] = (_Float16)in[i];
}

__global__ void k_fill(float* p, float v, long n) {
    long i = (long)blockIdx.x * blockDim.x + threadIdx.x;
    if (i < n) p[i] = v;
}

__global__ void k_pool_h4(const float* __restrict__ h4, float* __restrict__ out) {
    long i = (long)blockIdx.x * blockDim.x + threadIdx.x;   // 64*1024
    if (i >= (long)BSZ * 1024) return;
    long b = i >> 10, c = i & 1023;
    float s = 0.f;
    for (int n = 0; n < 49; ++n) s += h4[(b * 49 + n) * 1024 + c];
    out[i] = s * (1.f / 49.f);
}

__device__ inline void atomicMaxF(float* addr, float val) {
    unsigned* a = (unsigned*)addr;
    unsigned old = *a;
    while (__uint_as_float(old) < val) {
        unsigned assumed = old;
        old = atomicCAS(a, assumed, __float_as_uint(val));
        if (old == assumed) break;
    }
}

// wave per edge: e = att . leaky_relu(xl[src]+xr[dst]); segment max into m[dst]
__global__ void k_edge_scores(const float* __restrict__ xl,
                              const float* __restrict__ xr,
                              const float* __restrict__ att,
                              const int* __restrict__ src,
                              const int* __restrict__ dst, int E,
                              float* __restrict__ e_out, float* __restrict__ m) {
    int w = blockIdx.x * (blockDim.x >> 5) + (threadIdx.x >> 5);
    int lane = threadIdx.x & 31;
    if (w >= E) return;
    int s = src[w], d = dst[w];
    const float* ps = xl + (size_t)s * PROJ;
    const float* pd = xr + (size_t)d * PROJ;
    float acc = 0.f;
    for (int c = lane; c < PROJ; c += 32) {
        float v = ps[c] + pd[c];
        v = v > 0.f ? v : 0.2f * v;
        acc += att[c] * v;
    }
    for (int o = 16; o > 0; o >>= 1) acc += __shfl_down(acc, o, 32);
    if (lane == 0) {
        e_out[w] = acc;
        atomicMaxF(&m[d], acc);
    }
}

__global__ void k_edge_softmax(const float* __restrict__ e,
                               const float* __restrict__ m,
                               const int* __restrict__ dst, int E,
                               float* __restrict__ ex, float* __restrict__ denom) {
    int i = blockIdx.x * blockDim.x + threadIdx.x;
    if (i >= E) return;
    int d = dst[i];
    float v = expf(e[i] - m[d]);
    ex[i] = v;
    atomicAdd(&denom[d], v);
}

// wave per edge: agg[dst] += alpha * xl[src]
__global__ void k_edge_aggregate(const float* __restrict__ ex,
                                 const float* __restrict__ denom,
                                 const float* __restrict__ xl,
                                 const int* __restrict__ src,
                                 const int* __restrict__ dst, int E,
                                 float* __restrict__ agg) {
    int w = blockIdx.x * (blockDim.x >> 5) + (threadIdx.x >> 5);
    int lane = threadIdx.x & 31;
    if (w >= E) return;
    int s = src[w], d = dst[w];
    float alpha = ex[w] / denom[d];
    const float* ps = xl + (size_t)s * PROJ;
    float* pd = agg + (size_t)d * PROJ;
    for (int c = lane; c < PROJ; c += 32) atomicAdd(&pd[c], alpha * ps[c]);
}

__global__ void k_bias_act(const float* __restrict__ agg,
                           const float* __restrict__ bias,
                           float* __restrict__ x32, _Float16* __restrict__ x16,
                           long n, int act) {
    long i = (long)blockIdx.x * blockDim.x + threadIdx.x;
    if (i >= n) return;
    int c = (int)(i & (PROJ - 1));
    float v = agg[i] + bias[c];
    if (act == 2) v = v > 0.f ? v : 0.f;
    x32[i] = v;
    x16[i] = (_Float16)v;
}

__global__ void k_pool_nodes(const float* __restrict__ x, float* __restrict__ g) {
    int b = blockIdx.x;          // 64
    int c = threadIdx.x;         // 512
    float s = 0.f;
    for (int n = 0; n < NPG; ++n)
        s += x[((size_t)b * NPG + n) * PROJ + c];
    g[b * PROJ + c] = s * (1.f / (float)NPG);
}

// ---------------------------------------------------------------------------
// Host orchestration
// ---------------------------------------------------------------------------
static inline void gemm(hipStream_t st, const _Float16* A, const _Float16* B,
                        const float* bias, float* C32, _Float16* C16,
                        int M, int N, int K, int act,
                        int rpb = 0, int rbase = 0, int ldc = -1) {
    if (ldc < 0) ldc = N;
    dim3 grid((N + 127) / 128, (M + 127) / 128);
    wmma_gemm_kernel<<<grid, 256, 0, st>>>(A, B, bias, C32, C16, M, N, K, act,
                                           rpb, rbase, ldc);
}

static inline void tohalf(hipStream_t st, const float* in, _Float16* out, long n) {
    k_f32_to_f16<<<(int)((n + 255) / 256), 256, 0, st>>>(in, out, n);
}

extern "C" void kernel_launch(void* const* d_in, const int* in_sizes, int n_in,
                              void* d_out, int out_size, void* d_ws, size_t ws_size,
                              hipStream_t stream) {
    (void)n_in; (void)out_size; (void)ws_size;
    // ---- input indices (setup_inputs insertion order, leaves flattened) ----
    const float* h1 = (const float*)d_in[0];   // [64,784,256]
    const float* h2 = (const float*)d_in[1];   // [64,196,512]
    const float* h3 = (const float*)d_in[2];   // [64,49,1024]
    const float* h4 = (const float*)d_in[3];   // [64,49,1024]
    // proj0/1/2: (W0,b0,W1,b1,W2,b2) at 4..9 / 10..15 / 16..21
    const float* pW[3][3], *pb[3][3];
    for (int s = 0; s < 3; ++s)
        for (int l = 0; l < 3; ++l) {
            pW[s][l] = (const float*)d_in[4 + s * 6 + l * 2];
            pb[s][l] = (const float*)d_in[4 + s * 6 + l * 2 + 1];
        }
    // gat1/2/3: (Wl,bl,Wr,br,att,bias) at 22..27 / 28..33 / 34..39
    const float *gWl[3], *gbl[3], *gWr[3], *gbr[3], *gatt[3], *gbias[3];
    for (int g = 0; g < 3; ++g) {
        int o = 22 + g * 6;
        gWl[g]   = (const float*)d_in[o + 0];
        gbl[g]   = (const float*)d_in[o + 1];
        gWr[g]   = (const float*)d_in[o + 2];
        gbr[g]   = (const float*)d_in[o + 3];
        gatt[g]  = (const float*)d_in[o + 4];
        gbias[g] = (const float*)d_in[o + 5];
    }
    const float* linW = (const float*)d_in[40];
    const float* linb = (const float*)d_in[41];
    const int*   src  = (const int*)d_in[42];
    const int*   dst  = (const int*)d_in[43];
    const int    E    = in_sizes[42];

    // ---- workspace bump allocator ----
    char* base = (char*)d_ws;
    size_t off = 0;
    auto alloc = [&](size_t bytes) -> void* {
        void* p = base + off;
        off += (bytes + 255) & ~(size_t)255;
        return p;
    };
    const int pdims[3][4] = {{256, 1024, 256, PROJ},
                             {512, 1024, 256, PROJ},
                             {1024, 1024, 256, PROJ}};
    _Float16* pWh[3][3];
    for (int s = 0; s < 3; ++s)
        for (int l = 0; l < 3; ++l)
            pWh[s][l] = (_Float16*)alloc((size_t)pdims[s][l] * pdims[s][l + 1] * 2);
    _Float16 *gWlh[3], *gWrh[3];
    for (int g = 0; g < 3; ++g) {
        gWlh[g] = (_Float16*)alloc((size_t)PROJ * PROJ * 2);
        gWrh[g] = (_Float16*)alloc((size_t)PROJ * PROJ * 2);
    }
    _Float16* linWh = (_Float16*)alloc((size_t)PROJ * PROJ * 2);

    const long rows[4] = {BSZ * 784, BSZ * 196, BSZ * 49, BSZ};
    _Float16* h1h = (_Float16*)alloc((size_t)rows[0] * 256 * 2);
    _Float16* h2h = (_Float16*)alloc((size_t)rows[1] * 512 * 2);
    _Float16* h3h = (_Float16*)alloc((size_t)rows[2] * 1024 * 2);
    float*    pool32 = (float*)alloc((size_t)BSZ * 1024 * 4);
    _Float16* poolh  = (_Float16*)alloc((size_t)BSZ * 1024 * 2);
    _Float16* t1h = (_Float16*)alloc((size_t)rows[0] * 1024 * 2);
    _Float16* t2h = (_Float16*)alloc((size_t)rows[0] * 256 * 2);

    const long NX = (long)NNODE * PROJ;
    float*    x32  = (float*)alloc(NX * 4);
    _Float16* x16  = (_Float16*)alloc(NX * 2);
    float*    xl32 = (float*)alloc(NX * 4);
    float*    xr32 = (float*)alloc(NX * 4);
    float*    agg  = (float*)alloc(NX * 4);
    float*    ev   = (float*)alloc((size_t)E * 4);
    float*    exv  = (float*)alloc((size_t)E * 4);
    float*    mv   = (float*)alloc((size_t)NNODE * 4);
    float*    dv   = (float*)alloc((size_t)NNODE * 4);
    float*    g32  = (float*)alloc((size_t)BSZ * PROJ * 4);
    _Float16* g16  = (_Float16*)alloc((size_t)BSZ * PROJ * 2);

    // ---- 1. convert weights + inputs to f16 ----
    for (int s = 0; s < 3; ++s)
        for (int l = 0; l < 3; ++l)
            tohalf(stream, pW[s][l], pWh[s][l], (long)pdims[s][l] * pdims[s][l + 1]);
    for (int g = 0; g < 3; ++g) {
        tohalf(stream, gWl[g], gWlh[g], (long)PROJ * PROJ);
        tohalf(stream, gWr[g], gWrh[g], (long)PROJ * PROJ);
    }
    tohalf(stream, linW, linWh, (long)PROJ * PROJ);
    tohalf(stream, h1, h1h, rows[0] * 256);
    tohalf(stream, h2, h2h, rows[1] * 512);
    tohalf(stream, h3, h3h, rows[2] * 1024);
    k_pool_h4<<<(BSZ * 1024 + 255) / 256, 256, 0, stream>>>(h4, pool32);
    tohalf(stream, pool32, poolh, (long)BSZ * 1024);

    // ---- 2. projector MLP chains (gelu, gelu, linear) into x ----
    const _Float16* stageIn[4] = {h1h, h2h, h3h, poolh};
    const int stageProj[4] = {0, 1, 2, 2};  // pooled reuses proj2 (source bug)
    const int rbaseTab[4] = {0, 784, 980, 1029};
    const int rpbTab[4]   = {784, 196, 49, 1};
    for (int s = 0; s < 4; ++s) {
        int P0 = pdims[stageProj[s]][0];
        int M  = (int)rows[s];
        const _Float16* W0 = pWh[stageProj[s]][0];
        const _Float16* W1 = pWh[stageProj[s]][1];
        const _Float16* W2 = pWh[stageProj[s]][2];
        gemm(stream, stageIn[s], W0, pb[stageProj[s]][0], nullptr, t1h,
             M, 1024, P0, /*gelu*/1);
        gemm(stream, t1h, W1, pb[stageProj[s]][1], nullptr, t2h,
             M, 256, 1024, /*gelu*/1);
        gemm(stream, t2h, W2, pb[stageProj[s]][2], x32, x16,
             M, PROJ, 256, /*none*/0, rpbTab[s], rbaseTab[s], PROJ);
    }

    // ---- 3. three GATv2 layers ----
    const int EW = (E + 7) / 8;   // wave-per-edge grids (8 waves/block)
    for (int L = 0; L < 3; ++L) {
        gemm(stream, x16, gWlh[L], gbl[L], xl32, nullptr, NNODE, PROJ, PROJ, 0);
        gemm(stream, x16, gWrh[L], gbr[L], xr32, nullptr, NNODE, PROJ, PROJ, 0);
        k_fill<<<(NNODE + 255) / 256, 256, 0, stream>>>(mv, -1e30f, NNODE);
        k_fill<<<(NNODE + 255) / 256, 256, 0, stream>>>(dv, 0.f, NNODE);
        k_fill<<<(int)((NX + 255) / 256), 256, 0, stream>>>(agg, 0.f, NX);
        k_edge_scores<<<EW, 256, 0, stream>>>(xl32, xr32, gatt[L], src, dst, E,
                                              ev, mv);
        k_edge_softmax<<<(E + 255) / 256, 256, 0, stream>>>(ev, mv, dst, E,
                                                            exv, dv);
        k_edge_aggregate<<<EW, 256, 0, stream>>>(exv, dv, xl32, src, dst, E, agg);
        k_bias_act<<<(int)((NX + 255) / 256), 256, 0, stream>>>(
            agg, gbias[L], x32, x16, NX, (L < 2) ? 2 : 0);
    }

    // ---- 4. global mean pool + final linear into d_out ----
    k_pool_nodes<<<BSZ, PROJ, 0, stream>>>(x32, g32);
    tohalf(stream, g32, g16, (long)BSZ * PROJ);
    gemm(stream, g16, linWh, linb, (float*)d_out, nullptr, BSZ, PROJ, PROJ, 0);
}
